// ConvTM2d_75084618269072
// MI455X (gfx1250) — matrix operand order repaired and checked by reading, not compiled, not run on hardware
//
#include <hip/hip_runtime.h>

typedef __attribute__((ext_vector_type(16))) _Float16 v16h;
typedef __attribute__((ext_vector_type(8)))  _Float16 v8h;
typedef __attribute__((ext_vector_type(8)))  float    v8f;

#define F_IN   576
#define F_LIT  1152          // 2*F_IN literals
#define NCLS   256           // clauses
#define NOUT   128           // output classes
#define HW_    56
#define LSPAT  (HW_*HW_)     // 3136
#define NB     16
#define C_IN   64
#define KT1    36            // F_LIT/32 k-steps GEMM1
#define KT2    8             // NCLS/32 k-steps GEMM2

// ---- prep: include matrix -> binary f16 in WMMA B-fragment order -----------
// B-frag layout (16x16x32 f16, 32x16 B): half h of lane L holds
//   K = (h&7) + (h>=8 ? 16 : 0) + (L>=16 ? 8 : 0),  N = L&15.
// fragB[((kt*16 + nt)*32 + lane)*16 + h]
__global__ __launch_bounds__(256) void prep_include(const float* __restrict__ w,
                                                    _Float16* __restrict__ fragB) {
  int idx  = blockIdx.x * 256 + threadIdx.x;       // < 1152*256
  int h    = idx & 15;
  int lane = (idx >> 4) & 31;
  int fi   = idx >> 9;
  int kt   = fi >> 4;                              // 0..35
  int nt   = fi & 15;                              // 0..15
  int k = kt * 32 + (h & 7) + ((h & 8) ? 16 : 0) + ((lane & 16) ? 8 : 0);
  int n = nt * 16 + (lane & 15);
  // forward STE: include = (sigmoid(w) > 0.5) == (w > 0), exactly {0,1}
  fragB[idx] = (w[(size_t)n * F_LIT + k] > 0.0f) ? (_Float16)1.0f : (_Float16)0.0f;
}

// ---- prep: vote matrix -> f16 in WMMA B-fragment order ---------------------
// fragV[((kt*8 + nt)*32 + lane)*16 + h]
__global__ __launch_bounds__(256) void prep_vote(const float* __restrict__ vote,
                                                 _Float16* __restrict__ fragV) {
  int idx  = blockIdx.x * 256 + threadIdx.x;       // < 256*128
  int h    = idx & 15;
  int lane = (idx >> 4) & 31;
  int fi   = idx >> 9;
  int kt   = fi >> 3;                              // 0..7
  int nt   = fi & 7;                               // 0..7
  int k = kt * 32 + (h & 7) + ((h & 8) ? 16 : 0) + ((lane & 16) ? 8 : 0);
  int n = nt * 16 + (lane & 15);
  fragV[idx] = (_Float16)vote[(size_t)n * NCLS + k];
}

// ---- main fused kernel: im2col + log -> GEMM1 -> exp -> GEMM2 -> store -----
__global__ __launch_bounds__(256) void convtm_main(const float* __restrict__ x,
                                                   const _Float16* __restrict__ fragB,
                                                   const _Float16* __restrict__ fragV,
                                                   float* __restrict__ out) {
  __shared__ __align__(16) _Float16 sLog[16 * F_LIT];  // 36 KB: log-literals, 16 rows
  __shared__ __align__(16) _Float16 sCls[16 * NCLS];   //  8 KB: clause values, 16 rows

  const int tid  = threadIdx.x;
  const int lane = tid & 31;
  const int wave = tid >> 5;
  const int bi   = blockIdx.x / (LSPAT / 16);          // batch image
  const int pos0 = (blockIdx.x % (LSPAT / 16)) * 16;   // first spatial pos of tile

  // ---------- Phase A: fused unfold + literal logs into LDS ----------
  for (int it = 0; it < 36; ++it) {
    int idx = it * 256 + tid;                          // 0..9215 = 16 rows * 576 feats
    int r   = idx / F_IN;
    int f   = idx - r * F_IN;
    int c   = f / 9;
    int rem = f - c * 9;
    int khh = rem / 3;
    int kww = rem - khh * 3;
    int pos = pos0 + r;
    int oy  = pos / HW_;
    int ox  = pos - oy * HW_;
    int iy  = oy + khh - 1;                            // pad = 1
    int ix  = ox + kww - 1;
    float val = 0.0f;
    if (iy >= 0 && iy < HW_ && ix >= 0 && ix < HW_)
      val = x[(((size_t)bi * C_IN + c) * HW_ + iy) * HW_ + ix];
    sLog[r * F_LIT + f]        = (_Float16)__logf(val + 1e-6f);
    sLog[r * F_LIT + F_IN + f] = (_Float16)__logf(1.0f - val + 1e-6f);
  }
  __syncthreads();

  // A-fragment addressing (16x32 f16 A): lane holds M = lane&15;
  // halves 0..7 -> K = kbase..kbase+7, halves 8..15 -> K = kbase+16..kbase+23,
  // kbase = ks*32 + (lane>=16 ? 8 : 0).
  const int klo  = (lane & 16) ? 8 : 0;
  const int m    = lane & 15;
  const int mrow = (lane & 16) ? 8 : 0;                // C/D: M = vgpr + mrow

  // ---------- Phase B: GEMM1 over K=1152, each wave -> 2 clause tiles ----------
  const int nt0 = wave * 2;
  const int nt1 = nt0 + 1;
  v8f acc0 = {};
  v8f acc1 = {};
  for (int ks = 0; ks < KT1; ++ks) {
    const _Float16* ap = &sLog[m * F_LIT + ks * 32 + klo];
    v8h alo = *(const v8h*)(ap);
    v8h ahi = *(const v8h*)(ap + 16);
    v16h a  = __builtin_shufflevector(alo, ahi, 0, 1, 2, 3, 4, 5, 6, 7,
                                      8, 9, 10, 11, 12, 13, 14, 15);
    const _Float16* bp0 = &fragB[(((size_t)ks * 16 + nt0) * 32 + lane) * 16];
    const _Float16* bp1 = &fragB[(((size_t)ks * 16 + nt1) * 32 + lane) * 16];
    if (ks + 1 < KT1)
      __builtin_prefetch(&fragB[(((size_t)(ks + 1) * 16 + nt0) * 32 + lane) * 16], 0, 1);
    v16h b0 = *(const v16h*)(bp0);
    v16h b1 = *(const v16h*)(bp1);
    acc0 = __builtin_amdgcn_wmma_f32_16x16x32_f16(false, a, false, b0,
                                                  (short)0, acc0, false, false);
    acc1 = __builtin_amdgcn_wmma_f32_16x16x32_f16(false, a, false, b1,
                                                  (short)0, acc1, false, false);
  }
  // clauses = exp(acc); stash as f16 tile in LDS (C layout: row = v+mrow, col N = m)
#pragma unroll
  for (int vv = 0; vv < 8; ++vv) {
    sCls[(vv + mrow) * NCLS + nt0 * 16 + m] = (_Float16)__expf(acc0[vv]);
    sCls[(vv + mrow) * NCLS + nt1 * 16 + m] = (_Float16)__expf(acc1[vv]);
  }
  __syncthreads();

  // ---------- Phase C: GEMM2 over K=256, each wave -> 1 class tile ----------
  const int nt = wave;                                 // 8 waves x 16 = 128 classes
  v8f acc2 = {};
  for (int kt = 0; kt < KT2; ++kt) {
    const _Float16* ap = &sCls[m * NCLS + kt * 32 + klo];
    v8h alo = *(const v8h*)(ap);
    v8h ahi = *(const v8h*)(ap + 16);
    v16h a  = __builtin_shufflevector(alo, ahi, 0, 1, 2, 3, 4, 5, 6, 7,
                                      8, 9, 10, 11, 12, 13, 14, 15);
    v16h b  = *(const v16h*)(&fragV[(((size_t)kt * 8 + nt) * 32 + lane) * 16]);
    acc2 = __builtin_amdgcn_wmma_f32_16x16x32_f16(false, a, false, b,
                                                  (short)0, acc2, false, false);
  }

  // store logits: out[bi][cls][pos], cls = nt*16 + m, pos = pos0 + (vv + mrow)
  const size_t obase = ((size_t)bi * NOUT + (nt * 16 + m)) * LSPAT + pos0;
#pragma unroll
  for (int vv = 0; vv < 8; ++vv)
    out[obase + vv + mrow] = acc2[vv];
}

extern "C" void kernel_launch(void* const* d_in, const int* in_sizes, int n_in,
                              void* d_out, int out_size, void* d_ws, size_t ws_size,
                              hipStream_t stream) {
  const float* x         = (const float*)d_in[0];   // [16,64,56,56]
  const float* w_include = (const float*)d_in[1];   // [256,1152]
  const float* vote      = (const float*)d_in[2];   // [128,256]
  float*       out       = (float*)d_out;           // [16,128,56,56]

  _Float16* fragB = (_Float16*)d_ws;                        // 1152*256 f16 = 576 KB
  _Float16* fragV = fragB + (size_t)F_LIT * NCLS;           //  256*128 f16 =  64 KB

  prep_include<<<(F_LIT * NCLS) / 256, 256, 0, stream>>>(w_include, fragB);
  prep_vote<<<(NCLS * NOUT) / 256, 256, 0, stream>>>(vote, fragV);
  convtm_main<<<(NB * LSPAT) / 16, 256, 0, stream>>>(x, fragB, fragV, out);
}
